// InputMaskingNet_60112362275662
// MI455X (gfx1250) — matrix operand order repaired and checked by reading, not compiled, not run on hardware
//
#include <hip/hip_runtime.h>

typedef __attribute__((ext_vector_type(16))) __bf16 v16bf;
typedef __attribute__((ext_vector_type(8)))  float  v8f;

#define T_STEPS 64
#define B_ROWS  1024
#define D_DIM   512
#define BM      64           // batch rows per block
#define BN      64           // output cols per block
#define NTHR    256          // 8 waves (wave32)
#define LDSK    (D_DIM + 24) // 536 shorts = 1072 B = 268 dwords (67 x 16B): 16B-aligned rows,
                             // odd 16B stride -> per-phase bank-conflict-free

union FragBF16 { uint4 q[2]; unsigned u[8]; v16bf v; };

__device__ __forceinline__ unsigned short f32_bf16_rne(float f) {
  unsigned u = __float_as_uint(f);
  u += 0x7FFFu + ((u >> 16) & 1u);
  return (unsigned short)(u >> 16);
}
__device__ __forceinline__ float bf16_hi_f32(unsigned short h) {
  return __uint_as_float(((unsigned)h) << 16);
}

// Convert 8 consecutive floats into bf16 hi/lo packs (4 dwords each).
__device__ __forceinline__ void split8(const float* f, uint4& hv, uint4& lv) {
  unsigned hp[4], lp[4];
  #pragma unroll
  for (int j = 0; j < 4; ++j) {
    unsigned short h0 = f32_bf16_rne(f[2 * j]);
    unsigned short h1 = f32_bf16_rne(f[2 * j + 1]);
    unsigned short l0 = f32_bf16_rne(f[2 * j]     - bf16_hi_f32(h0));
    unsigned short l1 = f32_bf16_rne(f[2 * j + 1] - bf16_hi_f32(h1));
    hp[j] = (unsigned)h0 | ((unsigned)h1 << 16);
    lp[j] = (unsigned)l0 | ((unsigned)l1 << 16);
  }
  hv.x = hp[0]; hv.y = hp[1]; hv.z = hp[2]; hv.w = hp[3];
  lv.x = lp[0]; lv.y = lp[1]; lv.z = lp[2]; lv.w = lp[3];
}

__global__ __launch_bounds__(NTHR) void lif_linear_fused(
    const float* __restrict__ x, const float* __restrict__ W,
    const float* __restrict__ bias, float* __restrict__ out)
{
  extern __shared__ char smem[];
  unsigned short* xhi = (unsigned short*)smem;      // [BM][LDSK]
  unsigned short* xlo = xhi + BM * LDSK;            // [BM][LDSK]
  unsigned short* whi = xlo + BM * LDSK;            // [BN][LDSK]
  unsigned short* wlo = whi + BN * LDSK;            // [BN][LDSK]

  const int tid    = threadIdx.x;
  const int lane   = tid & 31;
  const int wave   = tid >> 5;       // 0..7
  const int waveM  = wave & 3;       // 4 x 16-row strips
  const int waveN  = wave >> 2;      // 2 x 32-col strips (2 tiles each)
  const bool hiHalf = (lane >= 16);
  const int hh8    = hiHalf ? 8 : 0;   // A-fragment K shift for upper lane half
  const int hh16   = hiHalf ? 16 : 0;  // unused: B fragments use hh8, same K order as A
  const int mn     = lane & 15;

  const int rowBase = blockIdx.x * BM;
  const int colBase = blockIdx.y * BN;

  // ---- Stage W tile (bf16 hi/lo split) once; reused for all 64 time steps ----
  for (int i = tid; i < (BN * D_DIM) / 8; i += NTHR) {
    int r = i >> 6;              // D_DIM/8 == 64 8-float chunks per row
    int c = (i & 63) << 3;
    const float4 a0 = *(const float4*)(W + (size_t)(colBase + r) * D_DIM + c);
    const float4 a1 = *(const float4*)(W + (size_t)(colBase + r) * D_DIM + c + 4);
    float f[8] = {a0.x, a0.y, a0.z, a0.w, a1.x, a1.y, a1.z, a1.w};
    uint4 hv, lv;
    split8(f, hv, lv);
    *(uint4*)(whi + r * LDSK + c) = hv;   // 16B-aligned -> ds_store_b128
    *(uint4*)(wlo + r * LDSK + c) = lv;
  }

  const int col0 = colBase + waveN * 32 + mn;
  const float bval0 = bias[col0];
  const float bval1 = bias[col0 + 16];
  v8f vstate0 = {}, vstate1 = {};   // membrane potentials, persistent across t

  const unsigned short* xh_row  = xhi + (size_t)(waveM * 16 + mn) * LDSK;
  const unsigned short* xl_row  = xlo + (size_t)(waveM * 16 + mn) * LDSK;
  const unsigned short* wh_row0 = whi + (size_t)(waveN * 32 + mn) * LDSK;
  const unsigned short* wl_row0 = wlo + (size_t)(waveN * 32 + mn) * LDSK;
  const unsigned short* wh_row1 = wh_row0 + (size_t)16 * LDSK;
  const unsigned short* wl_row1 = wl_row0 + (size_t)16 * LDSK;

  for (int t = 0; t < T_STEPS; ++t) {
    __syncthreads();   // previous compute done (and W staged, for t==0)

    // ---- Stage x[t] tile (bf16 hi/lo split); prefetch x[t+1] into L2 ----
    const float* xt = x + ((size_t)t * B_ROWS + rowBase) * D_DIM;
    for (int i = tid; i < (BM * D_DIM) / 8; i += NTHR) {
      int r = i >> 6;
      int c = (i & 63) << 3;
      const float4 a0 = *(const float4*)(xt + (size_t)r * D_DIM + c);
      const float4 a1 = *(const float4*)(xt + (size_t)r * D_DIM + c + 4);
      if (t < T_STEPS - 1) {  // uniform branch
        __builtin_prefetch(xt + (size_t)B_ROWS * D_DIM + (size_t)r * D_DIM + c, 0, 1);
      }
      float f[8] = {a0.x, a0.y, a0.z, a0.w, a1.x, a1.y, a1.z, a1.w};
      uint4 hv, lv;
      split8(f, hv, lv);
      *(uint4*)(xhi + r * LDSK + c) = hv;
      *(uint4*)(xlo + r * LDSK + c) = lv;
    }
    __syncthreads();

    // ---- K-loop: bf16x3 split-precision WMMA GEMM, 1x2 tiles per wave ----
    v8f acc0 = {}, acc1 = {};
    #pragma unroll 2
    for (int kk = 0; kk < D_DIM; kk += 32) {
      FragBF16 ah, al, bh0, bl0, bh1, bl1;
      // A 16x32 bf16: two contiguous 16B blocks per lane half (shared by both tiles)
      ah.q[0] = *(const uint4*)(xh_row + kk + hh8);
      ah.q[1] = *(const uint4*)(xh_row + kk + 16 + hh8);
      al.q[0] = *(const uint4*)(xl_row + kk + hh8);
      al.q[1] = *(const uint4*)(xl_row + kk + 16 + hh8);
      // B 32x16 bf16: two contiguous 16B blocks per lane half (same K order as A), two N-tiles
      bh0.q[0] = *(const uint4*)(wh_row0 + kk + hh8);
      bh0.q[1] = *(const uint4*)(wh_row0 + kk + 16 + hh8);
      bl0.q[0] = *(const uint4*)(wl_row0 + kk + hh8);
      bl0.q[1] = *(const uint4*)(wl_row0 + kk + 16 + hh8);
      bh1.q[0] = *(const uint4*)(wh_row1 + kk + hh8);
      bh1.q[1] = *(const uint4*)(wh_row1 + kk + 16 + hh8);
      bl1.q[0] = *(const uint4*)(wl_row1 + kk + hh8);
      bl1.q[1] = *(const uint4*)(wl_row1 + kk + 16 + hh8);

      acc0 = __builtin_amdgcn_wmma_f32_16x16x32_bf16(false, ah.v, false, bh0.v,
                                                     (short)0, acc0, false, false);
      acc1 = __builtin_amdgcn_wmma_f32_16x16x32_bf16(false, ah.v, false, bh1.v,
                                                     (short)0, acc1, false, false);
      acc0 = __builtin_amdgcn_wmma_f32_16x16x32_bf16(false, ah.v, false, bl0.v,
                                                     (short)0, acc0, false, false);
      acc1 = __builtin_amdgcn_wmma_f32_16x16x32_bf16(false, ah.v, false, bl1.v,
                                                     (short)0, acc1, false, false);
      acc0 = __builtin_amdgcn_wmma_f32_16x16x32_bf16(false, al.v, false, bh0.v,
                                                     (short)0, acc0, false, false);
      acc1 = __builtin_amdgcn_wmma_f32_16x16x32_bf16(false, al.v, false, bh1.v,
                                                     (short)0, acc1, false, false);
    }

    // ---- LIF update (tau=2, hard reset) + spike store, both N-tiles ----
    float* outt = out + (size_t)t * B_ROWS * D_DIM;
    #pragma unroll
    for (int r = 0; r < 8; ++r) {
      int M   = r + (hiHalf ? 8 : 0);      // C/D layout: VGPR r, lane half
      int row = rowBase + waveM * 16 + M;
      float h0  = acc0[r] + bval0;
      float vv0 = 0.5f * (vstate0[r] + h0);   // v + (h - v)/tau, tau = 2
      float s0  = (vv0 >= 1.0f) ? 1.0f : 0.0f;
      vstate0[r] = (vv0 >= 1.0f) ? 0.0f : vv0;
      outt[(size_t)row * D_DIM + col0] = s0;

      float h1  = acc1[r] + bval1;
      float vv1 = 0.5f * (vstate1[r] + h1);
      float s1  = (vv1 >= 1.0f) ? 1.0f : 0.0f;
      vstate1[r] = (vv1 >= 1.0f) ? 0.0f : vv1;
      outt[(size_t)row * D_DIM + col0 + 16] = s1;
    }
  }
}

extern "C" void kernel_launch(void* const* d_in, const int* in_sizes, int n_in,
                              void* d_out, int out_size, void* d_ws, size_t ws_size,
                              hipStream_t stream) {
  const float* x = (const float*)d_in[0];
  const float* W = (const float*)d_in[1];
  const float* b = (const float*)d_in[2];
  float* out = (float*)d_out;

  size_t smem = (size_t)(2 * BM + 2 * BN) * LDSK * sizeof(unsigned short); // 274 KB
  (void)hipFuncSetAttribute((const void*)lif_linear_fused,
                            hipFuncAttributeMaxDynamicSharedMemorySize,
                            (int)smem);
  dim3 grid(B_ROWS / BM, D_DIM / BN);   // 16 x 8 = 128 blocks
  hipLaunchKernelGGL(lif_linear_fused, grid, dim3(NTHR), smem, stream,
                     x, W, b, out);
}
